// TransformerCRF_80324478369840
// MI455X (gfx1250) — matrix operand order repaired and checked by reading, not compiled, not run on hardware
//
#include <hip/hip_runtime.h>
#include <hip/hip_bf16.h>
#include <math.h>

typedef __attribute__((ext_vector_type(8)))  _Float16 v8h;
typedef __attribute__((ext_vector_type(16))) _Float16 v16h;
typedef __attribute__((ext_vector_type(8)))  float    v8f;

#define B_   64
#define S_   512
#define D_   128
#define DFF_ 256
#define H_   4
#define T_   9
#define L_   2
#define HD_  32
#define M_   (B_*S_)   /* 32768 rows */

static __device__ __forceinline__ v16h cat8(v8h lo, v8h hi) {
  return __builtin_shufflevector(lo, hi, 0,1,2,3,4,5,6,7,8,9,10,11,12,13,14,15);
}
static __device__ __forceinline__ v8f wmma16(v16h a, v16h b, v8f c) {
  return __builtin_amdgcn_wmma_f32_16x16x32_f16(false, a, false, b, (short)0, c, false, false);
}
static __device__ __forceinline__ float hmax16(float v) {
  for (int m = 1; m < 16; m <<= 1) v = fmaxf(v, __shfl_xor(v, m, 32));
  return v;
}
static __device__ __forceinline__ float hsum16(float v) {
  for (int m = 1; m < 16; m <<= 1) v += __shfl_xor(v, m, 32);
  return v;
}
static __device__ __forceinline__ float wsum32(float v) {
  for (int m = 1; m < 32; m <<= 1) v += __shfl_xor(v, m, 32);
  return v;
}

// ---------------- f32 -> f16 weight convert ----------------
__global__ void cvt_f16_kernel(const float* __restrict__ in, _Float16* __restrict__ out, int n) {
  int i = blockIdx.x * 256 + threadIdx.x;
  if (i < n) out[i] = (_Float16)in[i];
}

// ---------------- embedding + positional encoding ----------------
__global__ void embed_pe_kernel(const int* __restrict__ sent, const float* __restrict__ emb,
                                float* __restrict__ xf32, _Float16* __restrict__ xf16,
                                float* __restrict__ keybias) {
  int idx = blockIdx.x * 256 + threadIdx.x;           // over M_*D_
  if (idx >= M_ * D_) return;
  int d   = idx & (D_ - 1);
  int row = idx >> 7;                                  // D_=128
  int s   = row & (S_ - 1);
  int tok = sent[row];
  float e = (tok == 0) ? 0.0f : emb[(size_t)tok * D_ + d];
  // div = exp(-(d&~1) * ln(10000)/128)
  float arg = (float)s * __expf(-(float)(d & ~1) * (9.210340371976184f / 128.0f));
  float pe  = (d & 1) ? __cosf(arg) : __sinf(arg);
  float v = e + pe;
  xf32[idx] = v;
  xf16[idx] = (_Float16)v;
  if (d == 0) keybias[row] = (tok == 0) ? -1.0e9f : 0.0f;
}

// ---------------- generic WMMA GEMM: out(MxN,f32) = A(MxK,f16) * W(NxK,f16)^T + bias ----------------
__global__ __launch_bounds__(32) void gemm_wmma_kernel(
    const _Float16* __restrict__ A, int lda,
    const _Float16* __restrict__ W, int ldb,
    const float* __restrict__ bias,
    float* __restrict__ out, int ldo, int K) {
  int tn = blockIdx.x, tm = blockIdx.y;
  int L  = threadIdx.x;
  int hf = L >> 4, lm = L & 15;
  const _Float16* arow = A + (size_t)(tm * 16 + lm) * lda + hf * 16;
  const _Float16* brow = W + (size_t)(tn * 16 + lm) * ldb + hf * 16;
  v8f c = {};
  for (int kk = 0; kk < K; kk += 32) {
    v8h alo = *(const v8h*)(arow + kk);
    v8h ahi = *(const v8h*)(arow + kk + 8);
    v16h a  = cat8(alo, ahi);
    v16h b  = *(const v16h*)(brow + kk);
    c = wmma16(a, b, c);
  }
  int col  = tn * 16 + lm;
  int row0 = tm * 16 + hf * 8;
  float bs = bias[col];
  for (int r = 0; r < 8; ++r)
    out[(size_t)(row0 + r) * ldo + col] = c[r] + bs;
}

// ---------------- split QKV (B,S,3D) f32 -> q,k (B,H,S,HD) f16, v transposed (B,H,HD,S) f16 ----------------
__global__ void split_qkv_kernel(const float* __restrict__ y,
                                 _Float16* __restrict__ q16, _Float16* __restrict__ k16,
                                 _Float16* __restrict__ vT16) {
  int idx = blockIdx.x * 256 + threadIdx.x;            // over M_*3*D_
  if (idx >= M_ * 3 * D_) return;
  int c   = idx % (3 * D_);
  int row = idx / (3 * D_);
  int which = c / D_;
  int within = c % D_;
  int h = within / HD_, hd = within % HD_;
  int b = row >> 9, s = row & (S_ - 1);
  _Float16 v = (_Float16)y[idx];
  int bh = b * H_ + h;
  if (which == 0)      q16[((size_t)bh * S_ + s) * HD_ + hd] = v;
  else if (which == 1) k16[((size_t)bh * S_ + s) * HD_ + hd] = v;
  else                 vT16[((size_t)bh * HD_ + hd) * S_ + s] = v;
}

// ---------------- fused (flash) attention: one wave per (b,h, 16-row q tile) ----------------
__global__ __launch_bounds__(32) void attn_kernel(
    const _Float16* __restrict__ q16, const _Float16* __restrict__ k16,
    const _Float16* __restrict__ vT16, const float* __restrict__ keybias,
    _Float16* __restrict__ ctx16) {
  int qt = blockIdx.x;
  int bh = blockIdx.y;
  int b = bh >> 2, h = bh & 3;
  int L = threadIdx.x;
  int hf = L >> 4, lm = L & 15;

  const _Float16* qb = q16 + (size_t)bh * S_ * HD_;
  const _Float16* kb = k16 + (size_t)bh * S_ * HD_;
  const _Float16* vb = vT16 + (size_t)bh * HD_ * S_;

  const _Float16* qrow = qb + (size_t)(qt * 16 + lm) * HD_ + hf * 16;
  v16h aq = cat8(*(const v8h*)qrow, *(const v8h*)(qrow + 8));

  float mrun[8], lrun[8];
  v8f c0 = {}, c1 = {};
  for (int r = 0; r < 8; ++r) { mrun[r] = -1.0e30f; lrun[r] = 0.0f; }

  __shared__ __align__(32) _Float16 pl[16 * 32];
  const float scale = 0.17677669529663687f;   // 1/sqrt(32)
  const float* kbias = keybias + b * S_;

  for (int k0 = 0; k0 < S_; k0 += 32) {
    v16h bk0 = *(const v16h*)(kb + (size_t)(k0 + lm) * HD_ + hf * 16);
    v16h bk1 = *(const v16h*)(kb + (size_t)(k0 + 16 + lm) * HD_ + hf * 16);
    v8f z = {};
    v8f s0 = wmma16(aq, bk0, z);
    v8f s1 = wmma16(aq, bk1, z);
    float kb0 = kbias[k0 + lm];
    float kb1 = kbias[k0 + 16 + lm];
    for (int r = 0; r < 8; ++r) {
      float a0 = s0[r] * scale + kb0;
      float a1 = s1[r] * scale + kb1;
      float mx = hmax16(fmaxf(a0, a1));
      float newm = fmaxf(mrun[r], mx);
      float alpha = __expf(mrun[r] - newm);
      float p0 = __expf(a0 - newm);
      float p1 = __expf(a1 - newm);
      float rs = hsum16(p0 + p1);
      lrun[r] = lrun[r] * alpha + rs;
      mrun[r] = newm;
      c0[r] *= alpha;
      c1[r] *= alpha;
      int prow = r + hf * 8;
      pl[prow * 32 + lm]      = (_Float16)p0;
      pl[prow * 32 + lm + 16] = (_Float16)p1;
    }
    // same-wave LDS ops are in-order; compiler inserts dscnt waits for the loads
    v16h ap = cat8(*(const v8h*)(&pl[lm * 32 + hf * 16]),
                   *(const v8h*)(&pl[lm * 32 + hf * 16 + 8]));
    v16h bv0 = *(const v16h*)(vb + (size_t)lm * S_ + k0 + hf * 16);
    v16h bv1 = *(const v16h*)(vb + (size_t)(lm + 16) * S_ + k0 + hf * 16);
    c0 = wmma16(ap, bv0, c0);
    c1 = wmma16(ap, bv1, c1);
  }
  for (int r = 0; r < 8; ++r) {
    float inv = 1.0f / lrun[r];
    int row = qt * 16 + r + hf * 8;
    size_t base = ((size_t)(b * S_ + row)) * D_ + h * HD_;
    ctx16[base + lm]      = (_Float16)(c0[r] * inv);
    ctx16[base + lm + 16] = (_Float16)(c1[r] * inv);
  }
}

// ---------------- residual add + LayerNorm (one wave per row of 128) ----------------
__global__ void add_ln_kernel(float* __restrict__ xf32, const float* __restrict__ y,
                              const float* __restrict__ g, const float* __restrict__ bb,
                              _Float16* __restrict__ xf16) {
  int wave = threadIdx.x >> 5, lane = threadIdx.x & 31;
  int row = blockIdx.x * 8 + wave;
  size_t base = (size_t)row * D_ + lane * 4;
  float v[4];
  float s = 0.0f;
  for (int i = 0; i < 4; ++i) { v[i] = xf32[base + i] + y[base + i]; s += v[i]; }
  s = wsum32(s);
  float mu = s * (1.0f / D_);
  float s2 = 0.0f;
  for (int i = 0; i < 4; ++i) { v[i] -= mu; s2 += v[i] * v[i]; }
  s2 = wsum32(s2);
  float rstd = rsqrtf(s2 * (1.0f / D_) + 1.0e-5f);
  for (int i = 0; i < 4; ++i) {
    float o = v[i] * rstd * g[lane * 4 + i] + bb[lane * 4 + i];
    xf32[base + i] = o;
    xf16[base + i] = (_Float16)o;
  }
}

// ---------------- ReLU + f16 convert ----------------
__global__ void relu_cvt_kernel(const float* __restrict__ in, _Float16* __restrict__ out, int n) {
  int i = blockIdx.x * 256 + threadIdx.x;
  if (i < n) out[i] = (_Float16)fmaxf(in[i], 0.0f);
}

// ---------------- emissions: (M,9) = x(M,128) * Wtag(9,128)^T + btag ----------------
__global__ void emis_kernel(const float* __restrict__ x, const float* __restrict__ Wtag,
                            const float* __restrict__ btag, float* __restrict__ em) {
  int idx = blockIdx.x * 256 + threadIdx.x;          // over M_*T_
  if (idx >= M_ * T_) return;
  int t = idx % T_, m = idx / T_;
  const float* xr = x + (size_t)m * D_;
  const float* wr = Wtag + (size_t)t * D_;
  float acc = btag[t];
  for (int d = 0; d < D_; ++d) acc += xr[d] * wr[d];
  em[idx] = acc;
}

// ---------------- CRF: forward logsumexp + gold score, mean over batch ----------------
__global__ void crf_kernel(const int* __restrict__ sent, const int* __restrict__ tags,
                           const float* __restrict__ em, const float* __restrict__ trans,
                           const float* __restrict__ start_t, const float* __restrict__ end_t,
                           float* __restrict__ out) {
  __shared__ float s_tr[T_ * T_];
  __shared__ float s_st[T_], s_en[T_];
  __shared__ float red[B_];
  int b = threadIdx.x;
  for (int i = b; i < T_ * T_; i += B_) s_tr[i] = trans[i];
  if (b < T_) { s_st[b] = start_t[b]; s_en[b] = end_t[b]; }
  __syncthreads();

  const float* e  = em + (size_t)b * S_ * T_;
  const int* sb   = sent + b * S_;
  const int* tb   = tags + b * S_;

  float fv[T_];
  for (int j = 0; j < T_; ++j) fv[j] = s_st[j] + e[j];
  for (int s = 1; s < S_; ++s) {
    if (sb[s] != 0) {
      const float* es = e + s * T_;
      float nv[T_];
      for (int j = 0; j < T_; ++j) {
        float mx = -1.0e30f;
        for (int i = 0; i < T_; ++i) mx = fmaxf(mx, fv[i] + s_tr[i * T_ + j]);
        float sm = 0.0f;
        for (int i = 0; i < T_; ++i) sm += __expf(fv[i] + s_tr[i * T_ + j] - mx);
        nv[j] = mx + __logf(sm) + es[j];
      }
      for (int j = 0; j < T_; ++j) fv[j] = nv[j];
    }
  }
  float mx = -1.0e30f;
  for (int j = 0; j < T_; ++j) mx = fmaxf(mx, fv[j] + s_en[j]);
  float sm = 0.0f;
  for (int j = 0; j < T_; ++j) sm += __expf(fv[j] + s_en[j] - mx);
  float fwd = mx + __logf(sm);

  int t0 = tb[0];
  float sc = s_st[t0] + e[t0];
  int seqlen = (sb[0] != 0) ? 1 : 0;
  int prev = t0;
  for (int s = 1; s < S_; ++s) {
    int cur = tb[s];
    if (sb[s] != 0) { sc += s_tr[prev * T_ + cur] + e[s * T_ + cur]; ++seqlen; }
    prev = cur;
  }
  if (seqlen < 1) seqlen = 1;
  sc += s_en[tb[seqlen - 1]];

  red[b] = fwd - sc;
  __syncthreads();
  if (b == 0) {
    float tot = 0.0f;
    for (int i = 0; i < B_; ++i) tot += red[i];
    out[0] = tot / (float)B_;
  }
}

extern "C" void kernel_launch(void* const* d_in, const int* in_sizes, int n_in,
                              void* d_out, int out_size, void* d_ws, size_t ws_size,
                              hipStream_t stream) {
  const int*   sent  = (const int*)d_in[0];
  const int*   tags  = (const int*)d_in[1];
  const float* emb   = (const float*)d_in[2];
  const float* Wqkv  = (const float*)d_in[3];
  const float* bqkv  = (const float*)d_in[4];
  const float* Wo    = (const float*)d_in[5];
  const float* bo    = (const float*)d_in[6];
  const float* W1    = (const float*)d_in[7];
  const float* b1f   = (const float*)d_in[8];
  const float* W2    = (const float*)d_in[9];
  const float* b2f   = (const float*)d_in[10];
  const float* ln1g  = (const float*)d_in[11];
  const float* ln1b  = (const float*)d_in[12];
  const float* ln2g  = (const float*)d_in[13];
  const float* ln2b  = (const float*)d_in[14];
  const float* Wtag  = (const float*)d_in[15];
  const float* btag  = (const float*)d_in[16];
  const float* trans = (const float*)d_in[17];
  const float* stt   = (const float*)d_in[18];
  const float* ent   = (const float*)d_in[19];

  char* w = (char*)d_ws;
  float*     xf32    = (float*)w;     w += (size_t)M_ * D_ * 4;
  _Float16*  xf16    = (_Float16*)w;  w += (size_t)M_ * D_ * 2;
  float*     keybias = (float*)w;     w += (size_t)B_ * S_ * 4;
  float*     y32     = (float*)w;     w += (size_t)M_ * 3 * D_ * 4;
  _Float16*  q16     = (_Float16*)w;  w += (size_t)M_ * D_ * 2;
  _Float16*  k16     = (_Float16*)w;  w += (size_t)M_ * D_ * 2;
  _Float16*  vT16    = (_Float16*)w;  w += (size_t)M_ * D_ * 2;
  _Float16*  ctx16   = (_Float16*)w;  w += (size_t)M_ * D_ * 2;
  _Float16*  ff116   = (_Float16*)w;  w += (size_t)M_ * DFF_ * 2;
  float*     em      = (float*)w;     w += (size_t)M_ * T_ * 4;
  _Float16*  Wqkv16  = (_Float16*)w;  w += (size_t)L_ * 3 * D_ * D_ * 2;
  _Float16*  Wo16    = (_Float16*)w;  w += (size_t)L_ * D_ * D_ * 2;
  _Float16*  W116    = (_Float16*)w;  w += (size_t)L_ * DFF_ * D_ * 2;
  _Float16*  W216    = (_Float16*)w;  w += (size_t)L_ * D_ * DFF_ * 2;

  // ---- weight conversions ----
  {
    int n;
    n = L_ * 3 * D_ * D_; cvt_f16_kernel<<<(n + 255) / 256, 256, 0, stream>>>(Wqkv, Wqkv16, n);
    n = L_ * D_ * D_;     cvt_f16_kernel<<<(n + 255) / 256, 256, 0, stream>>>(Wo, Wo16, n);
    n = L_ * DFF_ * D_;   cvt_f16_kernel<<<(n + 255) / 256, 256, 0, stream>>>(W1, W116, n);
    n = L_ * D_ * DFF_;   cvt_f16_kernel<<<(n + 255) / 256, 256, 0, stream>>>(W2, W216, n);
  }

  // ---- embedding + positional encoding ----
  {
    int n = M_ * D_;
    embed_pe_kernel<<<(n + 255) / 256, 256, 0, stream>>>(sent, emb, xf32, xf16, keybias);
  }

  for (int l = 0; l < L_; ++l) {
    // QKV: (M,384) = x(M,128) * Wqkv[l]^T + bqkv[l]
    gemm_wmma_kernel<<<dim3(3 * D_ / 16, M_ / 16), 32, 0, stream>>>(
        xf16, D_, Wqkv16 + (size_t)l * 3 * D_ * D_, D_, bqkv + (size_t)l * 3 * D_,
        y32, 3 * D_, D_);
    // split + permute
    {
      int n = M_ * 3 * D_;
      split_qkv_kernel<<<(n + 255) / 256, 256, 0, stream>>>(y32, q16, k16, vT16);
    }
    // fused attention -> ctx16 (B,S,D)
    attn_kernel<<<dim3(S_ / 16, B_ * H_), 32, 0, stream>>>(q16, k16, vT16, keybias, ctx16);
    // attn out projection
    gemm_wmma_kernel<<<dim3(D_ / 16, M_ / 16), 32, 0, stream>>>(
        ctx16, D_, Wo16 + (size_t)l * D_ * D_, D_, bo + (size_t)l * D_, y32, D_, D_);
    // x = LN(x + attnproj)
    add_ln_kernel<<<M_ / 8, 256, 0, stream>>>(xf32, y32, ln1g + (size_t)l * D_,
                                              ln1b + (size_t)l * D_, xf16);
    // FFN1: (M,256)
    gemm_wmma_kernel<<<dim3(DFF_ / 16, M_ / 16), 32, 0, stream>>>(
        xf16, D_, W116 + (size_t)l * DFF_ * D_, D_, b1f + (size_t)l * DFF_, y32, DFF_, D_);
    {
      int n = M_ * DFF_;
      relu_cvt_kernel<<<(n + 255) / 256, 256, 0, stream>>>(y32, ff116, n);
    }
    // FFN2: (M,128)
    gemm_wmma_kernel<<<dim3(D_ / 16, M_ / 16), 32, 0, stream>>>(
        ff116, DFF_, W216 + (size_t)l * D_ * DFF_, DFF_, b2f + (size_t)l * D_, y32, D_, DFF_);
    // x = LN(x + ff)
    add_ln_kernel<<<M_ / 8, 256, 0, stream>>>(xf32, y32, ln2g + (size_t)l * D_,
                                              ln2b + (size_t)l * D_, xf16);
  }

  // emissions
  {
    int n = M_ * T_;
    emis_kernel<<<(n + 255) / 256, 256, 0, stream>>>(xf32, Wtag, btag, em);
  }
  // CRF + mean
  crf_kernel<<<1, B_, 0, stream>>>(sent, tags, em, trans, stt, ent, (float*)d_out);
}
